// VQEncoder_25915832664381
// MI455X (gfx1250) — matrix operand order repaired and checked by reading, not compiled, not run on hardware
//
#include <hip/hip_runtime.h>
#include <stdint.h>

// Problem sizes (fixed by the reference)
#define BATCH 8
#define TSEQ  4096
#define CCH   1024
#define DCH   1024
#define KCB   2048
#define T2C   2048            // TSEQ / DS
#define M1    16384           // BATCH * T2
#define KIN   2048            // CCH * DS  (conv_in inner dim)

typedef __attribute__((ext_vector_type(16))) __bf16 v16bf;
typedef __attribute__((ext_vector_type(8)))  float  v8f;

union BfTile { v16bf v; uint4 q[2]; };

__device__ __forceinline__ unsigned short f32_to_bf16_bits(float f) {
  union { float f; uint32_t u; } cv; cv.f = f;
  uint32_t u = cv.u;
  u += 0x7FFFu + ((u >> 16) & 1u);   // round to nearest even
  return (unsigned short)(u >> 16);
}

__device__ __forceinline__ v8f wmma_bf16(const BfTile& a, const BfTile& b, v8f c) {
  return __builtin_amdgcn_wmma_f32_16x16x32_bf16(false, a.v, false, b.v, (short)0, c,
                                                 false, false);
}

__device__ __forceinline__ float block_reduce_sum_256(float v) {
  __shared__ float sm[256];
  int tid = threadIdx.x;
  sm[tid] = v; __syncthreads();
  for (int st = 128; st > 0; st >>= 1) {
    if (tid < st) sm[tid] += sm[tid + st];
    __syncthreads();
  }
  return sm[0];
}

// ---------------------------------------------------------------- converts
__global__ void k_cvt_x(const float* __restrict__ x, unsigned short* __restrict__ xb) {
  size_t i = ((size_t)blockIdx.x * blockDim.x + threadIdx.x) * 8;
  float4 f0 = *(const float4*)(x + i);
  float4 f1 = *(const float4*)(x + i + 4);
  uint4 o;
  o.x = (uint32_t)f32_to_bf16_bits(f0.x) | ((uint32_t)f32_to_bf16_bits(f0.y) << 16);
  o.y = (uint32_t)f32_to_bf16_bits(f0.z) | ((uint32_t)f32_to_bf16_bits(f0.w) << 16);
  o.z = (uint32_t)f32_to_bf16_bits(f1.x) | ((uint32_t)f32_to_bf16_bits(f1.y) << 16);
  o.w = (uint32_t)f32_to_bf16_bits(f1.z) | ((uint32_t)f32_to_bf16_bits(f1.w) << 16);
  *(uint4*)(xb + i) = o;
}

// W_in[d, c, s] -> Wfb[d, s*C + c]  (bf16), matching the contiguous 2C A-rows
__global__ void k_cvt_wf(const float* __restrict__ w, unsigned short* __restrict__ wfb) {
  int d  = blockIdx.y;
  int kk = blockIdx.x * blockDim.x + threadIdx.x;   // 0..2047
  int c  = kk & (CCH - 1);
  int s  = kk >> 10;
  wfb[(size_t)d * KIN + kk] = f32_to_bf16_bits(w[(size_t)d * KIN + c * 2 + s]);
}

__global__ void k_cvt_wo(const float* __restrict__ w, unsigned short* __restrict__ wob) {
  size_t i = (size_t)blockIdx.x * blockDim.x + threadIdx.x;
  wob[i] = f32_to_bf16_bits(w[i]);
}

// embed -> bf16 copy + per-code squared norm e2[k]
__global__ void k_embed_prep(const float* __restrict__ embed,
                             unsigned short* __restrict__ eb,
                             float* __restrict__ e2) {
  int k = blockIdx.x;
  const float* row = embed + (size_t)k * DCH;
  float s = 0.f;
  for (int d = threadIdx.x; d < DCH; d += 256) {
    float v = row[d];
    eb[(size_t)k * DCH + d] = f32_to_bf16_bits(v);
    s += v * v;
  }
  float tot = block_reduce_sum_256(s);
  if (threadIdx.x == 0) e2[k] = tot;
}

// ---------------------------------------------------------------- GEMM 1: z = x * Wf^T + b_in
// 32x64 per wave (2x4 WMMA tiles), 128x128 block tile, 8 waves.
__device__ __forceinline__ void store_z_tile(float* __restrict__ z,
                                             unsigned short* __restrict__ zb,
                                             const float* __restrict__ b_in,
                                             v8f acc, int mt, int nt, int l16, int lh) {
  int col = nt + l16;
  float bias = b_in[col];
  int rbase = mt + lh * 8;
  #pragma unroll
  for (int r = 0; r < 8; ++r) {
    float vo = acc[r] + bias;
    size_t off = (size_t)(rbase + r) * DCH + col;
    z[off]  = vo;
    zb[off] = f32_to_bf16_bits(vo);
  }
}

__global__ void __launch_bounds__(256, 1)
k_gemm_zin(const unsigned short* __restrict__ xb,
           const unsigned short* __restrict__ wfb,
           const float* __restrict__ b_in,
           float* __restrict__ z,
           unsigned short* __restrict__ zb) {
  const int lane = threadIdx.x & 31;
  const int wid  = threadIdx.x >> 5;
  const int l16  = lane & 15;
  const int lh   = lane >> 4;
  const int m0 = blockIdx.x * 128 + (wid & 3) * 32;   // 4 M-waves
  const int n0 = blockIdx.y * 128 + (wid >> 2) * 64;  // 2 N-waves
  const int ga = lh * 8;    // A: K groups {0..7,16..23} vs {8..15,24..31}
  const int gb = lh * 8;    // B: K groups {0..7,16..23} vs {8..15,24..31}

  // Single base pointer per operand; row steps are compile-time immediates.
  const unsigned short* apb = xb  + (size_t)(m0 + l16) * KIN;
  const unsigned short* bpb = wfb + (size_t)(n0 + l16) * KIN;

  v8f acc[2][4] = {};
  for (int kk = 0; kk < KIN; kk += 32) {
    BfTile a0, a1, b[4];
    a0.q[0] = *(const uint4*)(apb + kk + ga);
    a0.q[1] = *(const uint4*)(apb + kk + 16 + ga);
    a1.q[0] = *(const uint4*)(apb + 16 * KIN + kk + ga);
    a1.q[1] = *(const uint4*)(apb + 16 * KIN + kk + 16 + ga);
    #pragma unroll
    for (int j = 0; j < 4; ++j) {
      b[j].q[0] = *(const uint4*)(bpb + j * 16 * KIN + kk + gb);
      b[j].q[1] = *(const uint4*)(bpb + j * 16 * KIN + kk + 16 + gb);
    }
    #pragma unroll
    for (int j = 0; j < 4; ++j) {
      acc[0][j] = wmma_bf16(a0, b[j], acc[0][j]);
      acc[1][j] = wmma_bf16(a1, b[j], acc[1][j]);
    }
  }
  #pragma unroll
  for (int i = 0; i < 2; ++i)
    #pragma unroll
    for (int j = 0; j < 4; ++j)
      store_z_tile(z, zb, b_in, acc[i][j], m0 + i * 16, n0 + j * 16, l16, lh);
}

// ---------------------------------------------------------------- VQ distance GEMM + fused argmin
// 32 rows x 64 codes per wave pass (2x4 tiles).
__global__ void __launch_bounds__(256, 1)
k_vq_argmin(const unsigned short* __restrict__ zb,
            const unsigned short* __restrict__ eb,
            const float* __restrict__ e2,
            int* __restrict__ idx) {
  const int lane = threadIdx.x & 31;
  const int wid  = threadIdx.x >> 5;
  const int l16  = lane & 15;
  const int lh   = lane >> 4;
  const int m0   = (blockIdx.x * 8 + wid) * 32;   // 32 rows per wave

  const unsigned short* apb = zb + (size_t)(m0 + l16) * DCH;
  const int ga = lh * 8, gb = lh * 8;

  float best[2][8]; int bidx[2][8];
  #pragma unroll
  for (int t = 0; t < 2; ++t)
    #pragma unroll
    for (int r = 0; r < 8; ++r) { best[t][r] = 3.4e38f; bidx[t][r] = 0; }

  for (int n0 = 0; n0 < KCB; n0 += 64) {
    const unsigned short* bpb = eb + (size_t)(n0 + l16) * DCH;

    v8f acc[2][4] = {};
    for (int kk = 0; kk < DCH; kk += 32) {
      BfTile a0, a1, b[4];
      a0.q[0] = *(const uint4*)(apb + kk + ga);
      a0.q[1] = *(const uint4*)(apb + kk + 16 + ga);
      a1.q[0] = *(const uint4*)(apb + 16 * DCH + kk + ga);
      a1.q[1] = *(const uint4*)(apb + 16 * DCH + kk + 16 + ga);
      #pragma unroll
      for (int j = 0; j < 4; ++j) {
        b[j].q[0] = *(const uint4*)(bpb + j * 16 * DCH + kk + gb);
        b[j].q[1] = *(const uint4*)(bpb + j * 16 * DCH + kk + 16 + gb);
      }
      #pragma unroll
      for (int j = 0; j < 4; ++j) {
        acc[0][j] = wmma_bf16(a0, b[j], acc[0][j]);
        acc[1][j] = wmma_bf16(a1, b[j], acc[1][j]);
      }
    }
    #pragma unroll
    for (int j = 0; j < 4; ++j) {
      int   c   = n0 + j * 16 + l16;
      float e2c = e2[c];
      #pragma unroll
      for (int r = 0; r < 8; ++r) {
        float s0 = fmaf(-2.0f, acc[0][j][r], e2c);  // ||z||^2 row-constant: irrelevant
        float s1 = fmaf(-2.0f, acc[1][j][r], e2c);
        if (s0 < best[0][r]) { best[0][r] = s0; bidx[0][r] = c; }
        if (s1 < best[1][r]) { best[1][r] = s1; bidx[1][r] = c; }
      }
    }
  }
  // butterfly min-reduce across the 16 lanes holding the same rows (wave32 halves)
  #pragma unroll
  for (int mk = 1; mk < 16; mk <<= 1) {
    #pragma unroll
    for (int t = 0; t < 2; ++t)
      #pragma unroll
      for (int r = 0; r < 8; ++r) {
        float ov = __shfl_xor(best[t][r], mk, 32);
        int   oi = __shfl_xor(bidx[t][r], mk, 32);
        if (ov < best[t][r] || (ov == best[t][r] && oi < bidx[t][r])) {
          best[t][r] = ov; bidx[t][r] = oi;
        }
      }
  }
  if (l16 == 0) {
    #pragma unroll
    for (int t = 0; t < 2; ++t)
      #pragma unroll
      for (int r = 0; r < 8; ++r) idx[m0 + t * 16 + lh * 8 + r] = bidx[t][r];
  }
}

// ---------------------------------------------------------------- GEMM 3: out = W_out * quant (+dup x2) + b_out
__device__ __forceinline__ void store_out_tile(float* __restrict__ out,
                                               const float* __restrict__ b_out,
                                               v8f acc, int mt, int nt, int l16, int lh) {
  int col = nt + l16;
  float bias = b_out[col];
  int rbase = mt + lh * 8;
  #pragma unroll
  for (int r = 0; r < 8; ++r) {
    int m  = rbase + r;
    int b  = m >> 11;         // / T2
    int t2 = m & (T2C - 1);
    size_t base = (((size_t)b * TSEQ) + 2 * t2) * CCH + col;
    float vo = acc[r] + bias;
    out[base]       = vo;     // nearest-neighbor upsample x2
    out[base + CCH] = vo;
  }
}

__global__ void __launch_bounds__(256, 1)
k_gemm_out(const unsigned short* __restrict__ eb,
           const int* __restrict__ idx,
           const unsigned short* __restrict__ wob,
           const float* __restrict__ b_out,
           float* __restrict__ out) {
  const int lane = threadIdx.x & 31;
  const int wid  = threadIdx.x >> 5;
  const int l16  = lane & 15;
  const int lh   = lane >> 4;
  const int m0 = blockIdx.x * 128 + (wid & 3) * 32;
  const int n0 = blockIdx.y * 128 + (wid >> 2) * 64;
  const int ga = lh * 8, gb = lh * 8;

  // Gathered A rows = embed[idx] (two row-tiles -> two base pointers).
  const unsigned short* ap0 = eb + (size_t)idx[m0 + l16] * DCH;
  const unsigned short* ap1 = eb + (size_t)idx[m0 + 16 + l16] * DCH;
  const unsigned short* bpb = wob + (size_t)(n0 + l16) * DCH;

  v8f acc[2][4] = {};
  for (int kk = 0; kk < DCH; kk += 32) {
    BfTile a0, a1, b[4];
    a0.q[0] = *(const uint4*)(ap0 + kk + ga);
    a0.q[1] = *(const uint4*)(ap0 + kk + 16 + ga);
    a1.q[0] = *(const uint4*)(ap1 + kk + ga);
    a1.q[1] = *(const uint4*)(ap1 + kk + 16 + ga);
    #pragma unroll
    for (int j = 0; j < 4; ++j) {
      b[j].q[0] = *(const uint4*)(bpb + j * 16 * DCH + kk + gb);
      b[j].q[1] = *(const uint4*)(bpb + j * 16 * DCH + kk + 16 + gb);
    }
    #pragma unroll
    for (int j = 0; j < 4; ++j) {
      acc[0][j] = wmma_bf16(a0, b[j], acc[0][j]);
      acc[1][j] = wmma_bf16(a1, b[j], acc[1][j]);
    }
  }
  #pragma unroll
  for (int i = 0; i < 2; ++i)
    #pragma unroll
    for (int j = 0; j < 4; ++j)
      store_out_tile(out, b_out, acc[i][j], m0 + i * 16, n0 + j * 16, l16, lh);
}

// ---------------------------------------------------------------- loss = mean((embed[idx] - z)^2), deterministic 2-stage
__global__ void k_loss_partial(const float* __restrict__ z,
                               const float* __restrict__ embed,
                               const int* __restrict__ idx,
                               float* __restrict__ partial) {
  const size_t N = (size_t)M1 * DCH;
  float s = 0.f;
  for (size_t e = (size_t)blockIdx.x * blockDim.x + threadIdx.x; e < N;
       e += (size_t)gridDim.x * blockDim.x) {
    int m = (int)(e >> 10);
    int d = (int)(e & (DCH - 1));
    float q = embed[(size_t)idx[m] * DCH + d];
    float diff = q - z[e];
    s += diff * diff;
  }
  float tot = block_reduce_sum_256(s);
  if (threadIdx.x == 0) partial[blockIdx.x] = tot;
}

__global__ void k_loss_final(const float* __restrict__ partial, float* __restrict__ loss) {
  float s = 0.f;
  for (int i = threadIdx.x; i < 1024; i += 256) s += partial[i];
  float tot = block_reduce_sum_256(s);
  if (threadIdx.x == 0) *loss = tot * (1.0f / ((float)M1 * (float)DCH));
}

// ----------------------------------------------------------------
extern "C" void kernel_launch(void* const* d_in, const int* in_sizes, int n_in,
                              void* d_out, int out_size, void* d_ws, size_t ws_size,
                              hipStream_t stream) {
  (void)in_sizes; (void)n_in; (void)out_size; (void)ws_size;
  const float* x     = (const float*)d_in[0];
  const float* W_in  = (const float*)d_in[1];
  const float* b_in  = (const float*)d_in[2];
  const float* embed = (const float*)d_in[3];
  const float* W_out = (const float*)d_in[4];
  const float* b_out = (const float*)d_in[5];
  float* out  = (float*)d_out;
  float* loss = out + (size_t)BATCH * TSEQ * CCH;

  char* p = (char*)d_ws;
  auto take = [&](size_t bytes) { char* r = p; p += (bytes + 255) & ~(size_t)255; return r; };
  unsigned short* xb  = (unsigned short*)take((size_t)M1 * KIN * 2);   // 64 MB
  float*          z   = (float*)take((size_t)M1 * DCH * 4);            // 64 MB
  unsigned short* zb  = (unsigned short*)take((size_t)M1 * DCH * 2);   // 32 MB
  unsigned short* wfb = (unsigned short*)take((size_t)DCH * KIN * 2);  //  4 MB
  unsigned short* eb  = (unsigned short*)take((size_t)KCB * DCH * 2);  //  4 MB
  unsigned short* wob = (unsigned short*)take((size_t)CCH * DCH * 2);  //  2 MB
  float*          e2  = (float*)take((size_t)KCB * 4);
  int*            idx = (int*)take((size_t)M1 * 4);
  float*          prt = (float*)take(1024 * 4);

  // Stage 0: bf16 conversion / weight repack / codebook norms
  k_cvt_x     <<<(size_t)M1 * KIN / (256 * 8), 256, 0, stream>>>(x, xb);
  k_cvt_wf    <<<dim3(KIN / 256, DCH),         256, 0, stream>>>(W_in, wfb);
  k_embed_prep<<<KCB,                          256, 0, stream>>>(embed, eb, e2);
  k_cvt_wo    <<<(CCH * DCH) / 256,            256, 0, stream>>>(W_out, wob);

  // Stage 1: conv_in as WMMA GEMM -> z (f32 + bf16 copy); 128x128 block tile
  k_gemm_zin <<<dim3(M1 / 128, DCH / 128), 256, 0, stream>>>(xb, wfb, b_in, z, zb);

  // Stage 2: fused distance-GEMM + argmin -> idx; 32 rows/wave
  k_vq_argmin<<<M1 / 32 / 8, 256, 0, stream>>>(zb, eb, e2, idx);

  // Stage 3: conv_out as gathered WMMA GEMM, upsample x2 fused into stores
  k_gemm_out <<<dim3(M1 / 128, CCH / 128), 256, 0, stream>>>(eb, idx, wob, b_out, out);

  // Stage 4: commitment loss (deterministic two-stage reduction)
  k_loss_partial<<<1024, 256, 0, stream>>>(z, embed, idx, prt);
  k_loss_final  <<<1,    256, 0, stream>>>(prt, loss);
}
